// ONLSTM_60181081751948
// MI455X (gfx1250) — matrix operand order repaired and checked by reading, not compiled, not run on hardware
//
#include <hip/hip_runtime.h>
#include <hip/hip_fp16.h>
#include <hip/hip_bf16.h>

#define HDIM   1024
#define DDIM   512
#define TSTEPS 256
#define MB     64
#define NOUT   6144   // 6*H
#define KSPLIT 4

typedef __attribute__((ext_vector_type(16))) _Float16 v16h;
typedef __attribute__((ext_vector_type(8)))  float    v8f;

union Frag16 { uint4 u[2]; v16h v; };

#define WMMA_F16(acc, a, b)                                                    \
  acc = __builtin_amdgcn_wmma_f32_16x16x32_f16(false, (a).v, false, (b).v,     \
                                               (short)0, acc, false, false)

// ---------------------------------------------------------------- prep

__global__ void onlstm_zero_kernel(uint32_t* __restrict__ p, int n32) {
  int i = blockIdx.x * blockDim.x + threadIdx.x;
  if (i < n32) p[i] = 0u;
}

__global__ void onlstm_f32_to_f16_kernel(const float* __restrict__ in,
                                         _Float16* __restrict__ out, int n) {
  int i = blockIdx.x * blockDim.x + threadIdx.x;
  if (i < n) out[i] = (_Float16)in[i];
}

// Pre-swizzle W [K x 6144] f32 into per-lane WMMA B-fragment order (f16):
// tile (nt,kt): 32 lanes x 16 halves contiguous; lane -> (n = nt*16 + (lane&15),
// k-half = (lane<16 ? 0 : 16)); element e -> k = kt*32 + khalf + e.
__global__ void onlstm_swizzleW_kernel(const float* __restrict__ W,
                                       _Float16* __restrict__ out,
                                       int nkt, int total) {
  int i = blockIdx.x * blockDim.x + threadIdx.x;
  if (i >= total) return;
  int e    = i & 15;
  int lane = (i >> 4) & 31;
  int rem  = i >> 9;
  int kt   = rem % nkt;
  int nt   = rem / nkt;
  int k = kt * 32 + ((lane & 16) ? 16 : 0) + e;
  int n = nt * 16 + (lane & 15);
  out[i] = (_Float16)W[(size_t)k * NOUT + n];
}

// ---------------------------------------------------------------- GEMM

__device__ __forceinline__ void loadA_frag(Frag16& f, const _Float16* __restrict__ Ax,
                                           int Kx, const _Float16* __restrict__ Ah,
                                           int rowA, int klo, int ktG) {
  const int kfrag = ktG * 32 + klo;
  const _Float16* ap = (ktG * 32 < Kx)
      ? (Ax + (size_t)rowA * Kx + kfrag)
      : (Ah + (size_t)rowA * HDIM + (kfrag - Kx));
  f.u[0] = *(const uint4*)(ap);        // k: kfrag .. kfrag+7
  f.u[1] = *(const uint4*)(ap + 8);    // k: kfrag+8 .. kfrag+15
}

__device__ __forceinline__ void loadB_frag(Frag16& f, const _Float16* __restrict__ bp,
                                           int ktL) {
  const size_t ko = (size_t)ktL * 512;
  f.u[0] = *(const uint4*)(bp + ko);
  f.u[1] = *(const uint4*)(bp + ko + 8);
}

// Zp[ks][64 x 6144] partials = concat(Ax, Ah) @ W, K split KSPLIT ways.
// Wave: one 16x16 M-tile x 2 N-tiles x one K-split slice, double-buffered.
// 4 mt * 192 ng * 4 ks = 3072 waves -> 384 blocks x 256 threads.
__global__ __launch_bounds__(256) void onlstm_gemm_kernel(
    const _Float16* __restrict__ Ax, int Kx,
    const _Float16* __restrict__ Ah,
    const _Float16* __restrict__ Bsw,
    float* __restrict__ Zp, int K) {
  const int lane = threadIdx.x & 31;
  const int w    = blockIdx.x * 8 + (threadIdx.x >> 5);
  const int mt   = w & 3;            // M tile
  const int ks   = (w >> 2) & 3;     // K split slice
  const int ng   = w >> 4;           // 0..191 (pair of N tiles)
  const int nt0  = ng * 2;
  const int nkt  = K >> 5;
  const int nktL = nkt >> 2;         // k-steps per split (12 or 16)
  const int ktB  = ks * nktL;

  const int rowA = mt * 16 + (lane & 15);
  const int klo  = (lane < 16) ? 0 : 16;     // A-matrix k-half select

  const size_t bstride = (size_t)nkt * 512;  // halves per N-tile column
  const _Float16* bp0 =
      Bsw + (size_t)(nt0 + 0) * bstride + (size_t)ktB * 512 + (size_t)lane * 16;
  const _Float16* bp1 =
      Bsw + (size_t)(nt0 + 1) * bstride + (size_t)ktB * 512 + (size_t)lane * 16;

  float* Z = Zp + (size_t)ks * MB * NOUT;

  v8f acc0 = {}, acc1 = {};
  Frag16 a0, a1, b00, b01, b10, b11;

  loadA_frag(a0, Ax, Kx, Ah, rowA, klo, ktB);
  loadB_frag(b00, bp0, 0);
  loadB_frag(b01, bp1, 0);

  int kt = 0;
  for (; kt + 2 < nktL; kt += 2) {
    loadA_frag(a1, Ax, Kx, Ah, rowA, klo, ktB + kt + 1);
    loadB_frag(b10, bp0, kt + 1);
    loadB_frag(b11, bp1, kt + 1);
    if (kt + 6 < nktL) {
      __builtin_prefetch(bp0 + (size_t)(kt + 6) * 512, 0, 0);
      __builtin_prefetch(bp1 + (size_t)(kt + 6) * 512, 0, 0);
    }
    WMMA_F16(acc0, a0, b00);
    WMMA_F16(acc1, a0, b01);
    loadA_frag(a0, Ax, Kx, Ah, rowA, klo, ktB + kt + 2);
    loadB_frag(b00, bp0, kt + 2);
    loadB_frag(b01, bp1, kt + 2);
    WMMA_F16(acc0, a1, b10);
    WMMA_F16(acc1, a1, b11);
  }
  // tail: buf0 holds kt = nktL-2
  loadA_frag(a1, Ax, Kx, Ah, rowA, klo, ktB + nktL - 1);
  loadB_frag(b10, bp0, nktL - 1);
  loadB_frag(b11, bp1, nktL - 1);
  WMMA_F16(acc0, a0, b00);
  WMMA_F16(acc1, a0, b01);
  WMMA_F16(acc0, a1, b10);
  WMMA_F16(acc1, a1, b11);

  // C/D layout: VGPR r, lanes 0-15 -> M=r, lanes 16-31 -> M=r+8; N = lane&15
  const int rowC = mt * 16 + ((lane >> 4) << 3);
  const int col  = lane & 15;
#pragma unroll
  for (int r = 0; r < 8; ++r) {
    Z[(size_t)(rowC + r) * NOUT + (nt0 + 0) * 16 + col] = acc0[r];
    Z[(size_t)(rowC + r) * NOUT + (nt0 + 1) * 16 + col] = acc1[r];
  }
}

// ---------------------------------------------------------------- gates
// One block per batch row m (64 blocks, 256 threads, 4 hidden units/thread).
// Sums the KSPLIT Z-partials while reading. z order: i,f,g,o,ftilde,itilde.
__global__ __launch_bounds__(256) void onlstm_gate_kernel(
    const float* __restrict__ Zp, const float* __restrict__ bias,
    float* __restrict__ c, float* __restrict__ h,
    _Float16* __restrict__ h16,
    _Float16* __restrict__ extra16,  // optional second f16 copy of h (layer0 trace)
    float* __restrict__ outh) {      // optional f32 copy of h (layer1 all_h)
  const int m   = blockIdx.x;
  const int tid = threadIdx.x;
  const int j0  = tid * 4;
  const float* zr = Zp + (size_t)m * NOUT;
  const size_t ps = (size_t)MB * NOUT;
  __shared__ float red[256];

  auto zval = [&](int n) -> float {
    return zr[n] + zr[ps + n] + zr[2 * ps + n] + zr[3 * ps + n] + bias[n];
  };

  float ftil[4], itil[4];

  // two softmax+cumsum passes: chunk 4 (ftilde) then chunk 5 (itilde)
#pragma unroll
  for (int pass = 0; pass < 2; ++pass) {
    const int base = (pass == 0) ? 4 * HDIM : 5 * HDIM;
    float v[4];
#pragma unroll
    for (int e = 0; e < 4; ++e) v[e] = zval(base + j0 + e);
    float mx = fmaxf(fmaxf(v[0], v[1]), fmaxf(v[2], v[3]));
    red[tid] = mx; __syncthreads();
    for (int off = 128; off > 0; off >>= 1) {
      if (tid < off) red[tid] = fmaxf(red[tid], red[tid + off]);
      __syncthreads();
    }
    mx = red[0]; __syncthreads();

    float ex[4]; float s = 0.f;
#pragma unroll
    for (int e = 0; e < 4; ++e) { ex[e] = __expf(v[e] - mx); s += ex[e]; }
    red[tid] = s; __syncthreads();
    // Hillis-Steele inclusive scan of per-thread sums
    for (int off = 1; off < 256; off <<= 1) {
      float t = (tid >= off) ? red[tid - off] : 0.f;
      __syncthreads();
      red[tid] += t;
      __syncthreads();
    }
    const float total = red[255];
    const float excl  = red[tid] - s;
    __syncthreads();

    float run = excl;
#pragma unroll
    for (int e = 0; e < 4; ++e) {
      run += ex[e];
      float cum = run / total;
      if (pass == 0) ftil[e] = cum;
      else           itil[e] = 1.f - cum;
    }
  }

#pragma unroll
  for (int e = 0; e < 4; ++e) {
    const int j = j0 + e;
    const float zi = zval(0 * HDIM + j);
    const float zf = zval(1 * HDIM + j);
    const float zg = zval(2 * HDIM + j);
    const float zo = zval(3 * HDIM + j);
    const float it = 1.f / (1.f + __expf(-zi));
    const float ft = 1.f / (1.f + __expf(-zf));
    const float ot = 1.f / (1.f + __expf(-zo));
    const float ch = tanhf(zg);
    const float om = ftil[e] * itil[e];
    const float fh = ft * om + (ftil[e] - om);
    const float ih = it * om + (itil[e] - om);
    const size_t idx = (size_t)m * HDIM + j;
    const float cn = fh * c[idx] + ih * ch;
    const float hn = ot * tanhf(cn);
    c[idx]   = cn;
    h[idx]   = hn;
    h16[idx] = (_Float16)hn;
    if (extra16) extra16[idx] = (_Float16)hn;
    if (outh)    outh[idx]    = hn;
  }
}

__global__ void onlstm_finalize_kernel(const float* __restrict__ h0,
                                       const float* __restrict__ h1,
                                       const float* __restrict__ c0,
                                       const float* __restrict__ c1,
                                       float* __restrict__ out) {
  int i = blockIdx.x * blockDim.x + threadIdx.x;
  if (i < MB * HDIM) {
    out[i]                  = h0[i];
    out[MB * HDIM + i]      = h1[i];
    out[2 * MB * HDIM + i]  = c0[i];
    out[3 * MB * HDIM + i]  = c1[i];
  }
}

// ---------------------------------------------------------------- launch

extern "C" void kernel_launch(void* const* d_in, const int* in_sizes, int n_in,
                              void* d_out, int out_size, void* d_ws, size_t ws_size,
                              hipStream_t stream) {
  const float* input = (const float*)d_in[0];  // [T, M, D]
  const float* W0    = (const float*)d_in[1];  // [D+H, 6H]
  const float* b0    = (const float*)d_in[2];  // [6H]
  const float* W1    = (const float*)d_in[3];  // [2H, 6H]
  const float* b1    = (const float*)d_in[4];  // [6H]
  float* out = (float*)d_out;  // all_h [T,M,H] | h_n [2,M,H] | c_n [2,M,H]

  char* ws = (char*)d_ws;
  size_t off = 0;
  auto carve = [&](size_t bytes) -> void* {
    void* p = ws + off;
    off += (bytes + 255) & ~(size_t)255;
    return p;
  };
  const int K0 = DDIM + HDIM;   // 1536
  const int K1 = 2 * HDIM;      // 2048
  _Float16* W0sw = (_Float16*)carve((size_t)K0 * NOUT * 2);           // 18.9 MB
  _Float16* W1sw = (_Float16*)carve((size_t)K1 * NOUT * 2);           // 25.2 MB
  _Float16* Xf16 = (_Float16*)carve((size_t)TSTEPS * MB * DDIM * 2);  // 16.8 MB
  _Float16* H0tr = (_Float16*)carve((size_t)TSTEPS * MB * HDIM * 2);  // 33.6 MB
  float*    Zbuf = (float*)carve((size_t)KSPLIT * MB * NOUT * 4);     // 6.3 MB
  // contiguous state block (zeroed each call)
  char* stBase = (char*)carve(4 * (size_t)MB * HDIM * 4 + 2 * (size_t)MB * HDIM * 2);
  float*    h0f   = (float*)stBase;
  float*    c0f   = h0f + MB * HDIM;
  float*    h1f   = c0f + MB * HDIM;
  float*    c1f   = h1f + MB * HDIM;
  _Float16* h0f16 = (_Float16*)(c1f + MB * HDIM);
  _Float16* h1f16 = h0f16 + MB * HDIM;

  // 1) zero recurrent state (fp32 h/c + fp16 h)
  {
    int n32 = (4 * MB * HDIM * 4 + 2 * MB * HDIM * 2) / 4;
    onlstm_zero_kernel<<<(n32 + 255) / 256, 256, 0, stream>>>((uint32_t*)stBase, n32);
  }
  // 2) input -> fp16
  {
    int n = TSTEPS * MB * DDIM;
    onlstm_f32_to_f16_kernel<<<(n + 255) / 256, 256, 0, stream>>>(input, Xf16, n);
  }
  // 3) weight swizzle (fragment-order fp16)
  {
    int t0 = K0 * NOUT;
    onlstm_swizzleW_kernel<<<(t0 + 255) / 256, 256, 0, stream>>>(W0, W0sw, K0 / 32, t0);
    int t1 = K1 * NOUT;
    onlstm_swizzleW_kernel<<<(t1 + 255) / 256, 256, 0, stream>>>(W1, W1sw, K1 / 32, t1);
  }

  // 4) layer 0 recurrence
  for (int t = 0; t < TSTEPS; ++t) {
    const _Float16* Ax = Xf16 + (size_t)t * MB * DDIM;
    onlstm_gemm_kernel<<<384, 256, 0, stream>>>(Ax, DDIM, h0f16, W0sw, Zbuf, K0);
    onlstm_gate_kernel<<<MB, 256, 0, stream>>>(Zbuf, b0, c0f, h0f, h0f16,
                                               H0tr + (size_t)t * MB * HDIM, nullptr);
  }
  // 5) layer 1 recurrence (input = layer0 trace; fresh zero state per reference)
  for (int t = 0; t < TSTEPS; ++t) {
    const _Float16* Ax = H0tr + (size_t)t * MB * HDIM;
    onlstm_gemm_kernel<<<384, 256, 0, stream>>>(Ax, HDIM, h1f16, W1sw, Zbuf, K1);
    onlstm_gate_kernel<<<MB, 256, 0, stream>>>(Zbuf, b1, c1f, h1f, h1f16,
                                               nullptr, out + (size_t)t * MB * HDIM);
  }
  // 6) h_n / c_n tail
  onlstm_finalize_kernel<<<(MB * HDIM + 255) / 256, 256, 0, stream>>>(
      h0f, h1f, c0f, c1f, out + (size_t)TSTEPS * MB * HDIM);
}
